// MixedAttentionModule_82987358094251
// MI455X (gfx1250) — matrix operand order repaired and checked by reading, not compiled, not run on hardware
//
#include <hip/hip_runtime.h>
#include <cstdint>
#include <cstddef>

// ---------------- problem constants ----------------
#define BB   32
#define DD   384
#define HH   8
#define HD   48
#define NS   1024
#define NT   256
#define NN   1280            // NS + NT
#define FF   1536
#define KVMIX 320            // 256 search kv + 64 target kv
#define HDP  64              // head dim padded to multiple of 32 for f16 WMMA
#define MTOT (BB * NN)       // 40960 token rows
#define SCALE_Q 0.14433756729740643f  // 1/sqrt(48)

typedef __attribute__((ext_vector_type(16))) _Float16 v16h;
typedef __attribute__((ext_vector_type(8)))  float    v8f;
typedef __attribute__((ext_vector_type(4)))  unsigned int v4u;

// ---------------- WMMA helpers (gfx1250, wave32) ----------------
__device__ inline v8f wmma32(v16h a, v16h b, v8f c) {
  // D = A(16x32 f16) * B(32x16 f16) + C(16x16 f32)
  return __builtin_amdgcn_wmma_f32_16x16x32_f16(false, a, false, b, (short)0, c, false, false);
}

// A-matrix fragment (16x32 f16). rowptr points at this lane's row (M = lane&15),
// row-major over K. Lane halves cover K = hi*16..hi*16+15
// relative to k0 -> two contiguous 16B chunks.
__device__ inline v16h load_frag_a(const _Float16* rowptr, int k0, int hi) {
  union { v16h h; v4u u[2]; } f;
  const v4u* p = (const v4u*)(rowptr + k0);
  f.u[0] = p[2 * hi];
  f.u[1] = p[2 * hi + 1];
  return f.h;
}

// B-matrix fragment (32x16 f16). rowptr points at this lane's column's data,
// contiguous over K (i.e. row n of a row-major (N x K) matrix).
// Lane halves cover K = k0 + 16*hi .. +16 -> one contiguous 32B run.
__device__ inline v16h load_frag_b(const _Float16* rowptr, int k0, int hi) {
  union { v16h h; v4u u[2]; } f;
  const v4u* p = (const v4u*)(rowptr + k0 + hi * 16);
  f.u[0] = p[0];
  f.u[1] = p[1];
  return f.h;
}

// ---------------- elementwise: f32 -> f16 weight convert ----------------
__global__ void cvt_f32_f16(const float* __restrict__ in, _Float16* __restrict__ out, int n) {
  int i = blockIdx.x * 256 + threadIdx.x;
  if (i < n) out[i] = (_Float16)in[i];
}

// ---------------- LayerNorm 1: xn = LN(x) -> f16 ----------------
__global__ __launch_bounds__(128) void ln1_kernel(const float* __restrict__ x,
                                                  const float* __restrict__ g,
                                                  const float* __restrict__ b,
                                                  _Float16* __restrict__ out) {
  size_t t = blockIdx.x;
  const float* row = x + t * DD;
  int i0 = threadIdx.x;
  float v0 = row[i0], v1 = row[i0 + 128], v2 = row[i0 + 256];
  float s = v0 + v1 + v2;
  float s2 = v0 * v0 + v1 * v1 + v2 * v2;
  for (int m = 1; m < 32; m <<= 1) { s += __shfl_xor(s, m, 32); s2 += __shfl_xor(s2, m, 32); }
  __shared__ float red[8];
  int wave = threadIdx.x >> 5, lane = threadIdx.x & 31;
  if (lane == 0) { red[wave] = s; red[4 + wave] = s2; }
  __syncthreads();
  s = red[0] + red[1] + red[2] + red[3];
  s2 = red[4] + red[5] + red[6] + red[7];
  float mean = s * (1.0f / DD);
  float var = s2 * (1.0f / DD) - mean * mean;
  float rstd = rsqrtf(var + 1e-5f);
  _Float16* orow = out + t * DD;
  orow[i0]       = (_Float16)((v0 - mean) * rstd * g[i0]       + b[i0]);
  orow[i0 + 128] = (_Float16)((v1 - mean) * rstd * g[i0 + 128] + b[i0 + 128]);
  orow[i0 + 256] = (_Float16)((v2 - mean) * rstd * g[i0 + 256] + b[i0 + 256]);
}

// ---------------- residual + LayerNorm 2: y = x + ao (f32), h = LN(y) (f16) ----
__global__ __launch_bounds__(128) void ln2_kernel(const float* __restrict__ x,
                                                  const float* __restrict__ ao,
                                                  const float* __restrict__ g,
                                                  const float* __restrict__ b,
                                                  float* __restrict__ y,
                                                  _Float16* __restrict__ out) {
  size_t t = blockIdx.x;
  const float* xr = x + t * DD;
  const float* ar = ao + t * DD;
  float* yr = y + t * DD;
  int i0 = threadIdx.x;
  float v0 = xr[i0] + ar[i0];
  float v1 = xr[i0 + 128] + ar[i0 + 128];
  float v2 = xr[i0 + 256] + ar[i0 + 256];
  yr[i0] = v0; yr[i0 + 128] = v1; yr[i0 + 256] = v2;
  float s = v0 + v1 + v2;
  float s2 = v0 * v0 + v1 * v1 + v2 * v2;
  for (int m = 1; m < 32; m <<= 1) { s += __shfl_xor(s, m, 32); s2 += __shfl_xor(s2, m, 32); }
  __shared__ float red[8];
  int wave = threadIdx.x >> 5, lane = threadIdx.x & 31;
  if (lane == 0) { red[wave] = s; red[4 + wave] = s2; }
  __syncthreads();
  s = red[0] + red[1] + red[2] + red[3];
  s2 = red[4] + red[5] + red[6] + red[7];
  float mean = s * (1.0f / DD);
  float var = s2 * (1.0f / DD) - mean * mean;
  float rstd = rsqrtf(var + 1e-5f);
  _Float16* orow = out + t * DD;
  orow[i0]       = (_Float16)((v0 - mean) * rstd * g[i0]       + b[i0]);
  orow[i0 + 128] = (_Float16)((v1 - mean) * rstd * g[i0 + 128] + b[i0 + 128]);
  orow[i0 + 256] = (_Float16)((v2 - mean) * rstd * g[i0 + 256] + b[i0 + 256]);
}

// ---------------- depthwise 3x3 conv + BN (tokens are (hw,hw,D)) ----------------
__global__ void dwconv_kernel(const _Float16* __restrict__ xn, const float* __restrict__ w,
                              const float* __restrict__ cb, const float* __restrict__ g,
                              const float* __restrict__ bb, _Float16* __restrict__ out,
                              int in_off, int ihw, int out_off, int ohw, int stride,
                              int MoutTotal, int n_out) {
  int idx = blockIdx.x * 256 + threadIdx.x;
  if (idx >= n_out) return;
  int d = idx % DD;
  int t = (idx / DD) % (ohw * ohw);
  int b = idx / (DD * ohw * ohw);
  int ro = t / ohw, co = t % ohw;
  const _Float16* base = xn + ((size_t)b * NN + in_off) * DD + d;
  float acc = 0.0f;
#pragma unroll
  for (int i = 0; i < 3; ++i) {
    int r = ro * stride + i - 1;
    if (r < 0 || r >= ihw) continue;
#pragma unroll
    for (int j = 0; j < 3; ++j) {
      int c = co * stride + j - 1;
      if (c < 0 || c >= ihw) continue;
      acc += (float)base[(size_t)(r * ihw + c) * DD] * w[d * 9 + i * 3 + j];
    }
  }
  float sc = g[d] * rsqrtf(1.0f + 1e-5f);
  float val = (acc + cb[d]) * sc + bb[d];
  out[((size_t)b * MoutTotal + out_off + t) * DD + d] = (_Float16)val;
}

// ---------------- WMMA GEMM: projection (out = A @ W^T + bias, head-split write) --
// A: (Mtot x 384) f16 row-major. W: (384 x 384) f16 row-major over K.
// mode 0: q -> qp[(b*H+h)*Mper + m][d] (64-padded), scaled
// mode 1: k -> kp  (same layout, Mper = 320)
// mode 2: v -> vt[(b*H+h)*64 + d][m]  (transposed for P*V B-fragments)
__global__ __launch_bounds__(256) void gemm_proj(const _Float16* __restrict__ A,
                                                 const _Float16* __restrict__ W,
                                                 const float* __restrict__ bias,
                                                 _Float16* __restrict__ dst,
                                                 int Mper, int mode, float scale) {
  int tid = threadIdx.x, wave = tid >> 5, lane = tid & 31;
  int hi = lane >> 4, ln = lane & 15;
  int row0 = blockIdx.x * 128 + wave * 16;
  int nb = blockIdx.y * 64;
  const _Float16* arow = A + (size_t)(row0 + ln) * DD;
  v8f acc[4] = {v8f{}, v8f{}, v8f{}, v8f{}};
  for (int k0 = 0; k0 < DD; k0 += 32) {
    v16h af = load_frag_a(arow, k0, hi);
#pragma unroll
    for (int t = 0; t < 4; ++t) {
      v16h bf = load_frag_b(W + (size_t)(nb + t * 16 + ln) * DD, k0, hi);
      acc[t] = wmma32(af, bf, acc[t]);
    }
  }
#pragma unroll
  for (int t = 0; t < 4; ++t) {
    int n = nb + t * 16 + ln;
    int head = n / HD, d = n - head * HD;
#pragma unroll
    for (int r = 0; r < 8; ++r) {
      int m = row0 + r + 8 * hi;
      int b = m / Mper, mtok = m - b * Mper;
      float v = (acc[t][r] + bias[n]) * scale;
      if (mode == 2) {
        dst[((size_t)(b * HH + head) * HDP + d) * KVMIX + mtok] = (_Float16)v;
      } else {
        dst[((size_t)(b * HH + head) * Mper + mtok) * HDP + d] = (_Float16)v;
      }
    }
  }
}

// ---------------- WMMA GEMM: FF1 + exact GELU -> f16 ----------------
__global__ __launch_bounds__(256) void gemm_ff1(const _Float16* __restrict__ A,
                                                const _Float16* __restrict__ W,
                                                const float* __restrict__ bias,
                                                _Float16* __restrict__ gout) {
  int tid = threadIdx.x, wave = tid >> 5, lane = tid & 31;
  int hi = lane >> 4, ln = lane & 15;
  int row0 = blockIdx.x * 128 + wave * 16;
  int nb = blockIdx.y * 64;
  const _Float16* arow = A + (size_t)(row0 + ln) * DD;
  v8f acc[4] = {v8f{}, v8f{}, v8f{}, v8f{}};
  for (int k0 = 0; k0 < DD; k0 += 32) {
    v16h af = load_frag_a(arow, k0, hi);
#pragma unroll
    for (int t = 0; t < 4; ++t) {
      v16h bf = load_frag_b(W + (size_t)(nb + t * 16 + ln) * DD, k0, hi);
      acc[t] = wmma32(af, bf, acc[t]);
    }
  }
#pragma unroll
  for (int t = 0; t < 4; ++t) {
    int n = nb + t * 16 + ln;
    float bn = bias[n];
#pragma unroll
    for (int r = 0; r < 8; ++r) {
      int m = row0 + r + 8 * hi;
      float v = acc[t][r] + bn;
      float ge = 0.5f * v * (1.0f + erff(v * 0.70710678118f));
      gout[(size_t)m * FF + n] = (_Float16)ge;
    }
  }
}

// ---------------- WMMA GEMM: FF2 + bias + residual -> d_out (f32) ----------------
__global__ __launch_bounds__(256) void gemm_ff2(const _Float16* __restrict__ A,
                                                const _Float16* __restrict__ W,
                                                const float* __restrict__ bias,
                                                const float* __restrict__ y,
                                                float* __restrict__ out) {
  int tid = threadIdx.x, wave = tid >> 5, lane = tid & 31;
  int hi = lane >> 4, ln = lane & 15;
  int row0 = blockIdx.x * 128 + wave * 16;
  int nb = blockIdx.y * 64;
  const _Float16* arow = A + (size_t)(row0 + ln) * FF;
  v8f acc[4] = {v8f{}, v8f{}, v8f{}, v8f{}};
  for (int k0 = 0; k0 < FF; k0 += 32) {
    v16h af = load_frag_a(arow, k0, hi);
#pragma unroll
    for (int t = 0; t < 4; ++t) {
      v16h bf = load_frag_b(W + (size_t)(nb + t * 16 + ln) * FF, k0, hi);
      acc[t] = wmma32(af, bf, acc[t]);
    }
  }
#pragma unroll
  for (int t = 0; t < 4; ++t) {
    int n = nb + t * 16 + ln;
    float bn = bias[n];
#pragma unroll
    for (int r = 0; r < 8; ++r) {
      int m = row0 + r + 8 * hi;
      out[(size_t)m * DD + n] = acc[t][r] + bn + y[(size_t)m * DD + n];
    }
  }
}

// ---------------- flash attention (one wave = one 16-row q tile) ----------------
// qp: (B*H, NN, 64)  kp: (B*H, KVMIX, 64)  vt: (B*H, 64, KVMIX)  ao: (B, NN, 384)
__global__ __launch_bounds__(128) void attn_kernel(const _Float16* __restrict__ qp,
                                                   const _Float16* __restrict__ kp,
                                                   const _Float16* __restrict__ vt,
                                                   float* __restrict__ ao,
                                                   int q_base, int kv_base, int kv_len) {
  __shared__ _Float16 sh[4][16 * 32];   // per-wave P-tile transpose buffer
  int tid = threadIdx.x, wave = tid >> 5, lane = tid & 31;
  int hi = lane >> 4, ln = lane & 15;
  int bh = blockIdx.x;
  int b = bh >> 3, h = bh & 7;
  int qrow0 = q_base + (blockIdx.y * 4 + wave) * 16;

  const _Float16* qrow = qp + ((size_t)bh * NN + qrow0 + ln) * HDP;
  v16h qa0 = load_frag_a(qrow, 0, hi);
  v16h qa1 = load_frag_a(qrow, 32, hi);

  float rmax[8], rsum[8];
#pragma unroll
  for (int r = 0; r < 8; ++r) { rmax[r] = -3.0e38f; rsum[r] = 0.0f; }
  v8f o0 = v8f{}, o1 = v8f{}, o2 = v8f{};

  for (int c = 0; c < kv_len; c += 32) {
    int kv0 = kv_base + c;
    // scores: two 16x16 tiles over 32 kv cols, K = 64 (padded head dim)
    const _Float16* kr0 = kp + ((size_t)bh * KVMIX + kv0 + ln) * HDP;
    const _Float16* kr1 = kp + ((size_t)bh * KVMIX + kv0 + 16 + ln) * HDP;
    v8f s0 = v8f{}, s1 = v8f{};
    s0 = wmma32(qa0, load_frag_b(kr0, 0, hi), s0);
    s0 = wmma32(qa1, load_frag_b(kr0, 32, hi), s0);
    s1 = wmma32(qa0, load_frag_b(kr1, 0, hi), s1);
    s1 = wmma32(qa1, load_frag_b(kr1, 32, hi), s1);

    float alpha[8];
#pragma unroll
    for (int r = 0; r < 8; ++r) {
      float cm = fmaxf(s0[r], s1[r]);
      cm = fmaxf(cm, __shfl_xor(cm, 1, 32));
      cm = fmaxf(cm, __shfl_xor(cm, 2, 32));
      cm = fmaxf(cm, __shfl_xor(cm, 4, 32));
      cm = fmaxf(cm, __shfl_xor(cm, 8, 32));
      float nm = fmaxf(rmax[r], cm);
      alpha[r] = __expf(rmax[r] - nm);
      rmax[r] = nm;
      float p0 = __expf(s0[r] - nm);
      float p1 = __expf(s1[r] - nm);
      s0[r] = p0; s1[r] = p1;
      float rs = p0 + p1;
      rs += __shfl_xor(rs, 1, 32);
      rs += __shfl_xor(rs, 2, 32);
      rs += __shfl_xor(rs, 4, 32);
      rs += __shfl_xor(rs, 8, 32);
      rsum[r] = rsum[r] * alpha[r] + rs;
    }
#pragma unroll
    for (int r = 0; r < 8; ++r) { o0[r] *= alpha[r]; o1[r] *= alpha[r]; o2[r] *= alpha[r]; }

    // transpose P (C layout) -> A layout via wave-private LDS
    _Float16* s = sh[wave];
#pragma unroll
    for (int r = 0; r < 8; ++r) {
      int m = r + 8 * hi;
      s[m * 32 + ln]      = (_Float16)s0[r];
      s[m * 32 + 16 + ln] = (_Float16)s1[r];
    }
    v16h pa = load_frag_a(sh[wave] + ln * 32, 0, hi);

    // O += P (16x32) * V (32x48): V transposed so B-fragments are contiguous
    v16h vb0 = load_frag_b(vt + ((size_t)bh * HDP + 0 + ln) * KVMIX, kv0, hi);
    v16h vb1 = load_frag_b(vt + ((size_t)bh * HDP + 16 + ln) * KVMIX, kv0, hi);
    v16h vb2 = load_frag_b(vt + ((size_t)bh * HDP + 32 + ln) * KVMIX, kv0, hi);
    o0 = wmma32(pa, vb0, o0);
    o1 = wmma32(pa, vb1, o1);
    o2 = wmma32(pa, vb2, o2);
  }

#pragma unroll
  for (int r = 0; r < 8; ++r) {
    float inv = 1.0f / rsum[r];
    int m = r + 8 * hi;
    int token = qrow0 + m;
    float* dst = ao + ((size_t)b * NN + token) * DD + h * HD;
    dst[ln]      = o0[r] * inv;
    dst[16 + ln] = o1[r] * inv;
    dst[32 + ln] = o2[r] * inv;
  }
}

// ---------------- host launcher ----------------
extern "C" void kernel_launch(void* const* d_in, const int* in_sizes, int n_in,
                              void* d_out, int out_size, void* d_ws, size_t ws_size,
                              hipStream_t stream) {
  (void)in_sizes; (void)n_in; (void)out_size; (void)ws_size;
  const float* x     = (const float*)d_in[0];
  const float* ln1_g = (const float*)d_in[1];
  const float* ln1_b = (const float*)d_in[2];
  const float* dwq_w = (const float*)d_in[3];
  const float* dwq_b = (const float*)d_in[4];
  const float* bnq_g = (const float*)d_in[5];
  const float* bnq_b = (const float*)d_in[6];
  const float* dwk_w = (const float*)d_in[7];
  const float* dwk_b = (const float*)d_in[8];
  const float* bnk_g = (const float*)d_in[9];
  const float* bnk_b = (const float*)d_in[10];
  const float* dwv_w = (const float*)d_in[11];
  const float* dwv_b = (const float*)d_in[12];
  const float* bnv_g = (const float*)d_in[13];
  const float* bnv_b = (const float*)d_in[14];
  const float* pq_w  = (const float*)d_in[15];
  const float* pq_b  = (const float*)d_in[16];
  const float* pk_w  = (const float*)d_in[17];
  const float* pk_b  = (const float*)d_in[18];
  const float* pv_w  = (const float*)d_in[19];
  const float* pv_b  = (const float*)d_in[20];
  const float* ln2_g = (const float*)d_in[21];
  const float* ln2_b = (const float*)d_in[22];
  const float* ff1_w = (const float*)d_in[23];
  const float* ff1_b = (const float*)d_in[24];
  const float* ff2_w = (const float*)d_in[25];
  const float* ff2_b = (const float*)d_in[26];
  float* out = (float*)d_out;

  // workspace carve-up (256B aligned)
  char* base = (char*)d_ws;
  size_t off = 0;
  auto take = [&](size_t bytes) -> char* {
    off = (off + 255) & ~(size_t)255;
    char* p = base + off;
    off += bytes;
    return p;
  };
  _Float16* xn  = (_Float16*)take((size_t)MTOT * DD * 2);
  _Float16* qc  = (_Float16*)take((size_t)MTOT * DD * 2);
  _Float16* kc  = (_Float16*)take((size_t)BB * KVMIX * DD * 2);
  _Float16* vc  = (_Float16*)take((size_t)BB * KVMIX * DD * 2);
  _Float16* qpb = (_Float16*)take((size_t)BB * HH * NN * HDP * 2);
  _Float16* kpb = (_Float16*)take((size_t)BB * HH * KVMIX * HDP * 2);
  _Float16* vtb = (_Float16*)take((size_t)BB * HH * HDP * KVMIX * 2);
  float*    ao  = (float*)take((size_t)MTOT * DD * 4);
  float*    y   = (float*)take((size_t)MTOT * DD * 4);
  _Float16* hh  = (_Float16*)take((size_t)MTOT * DD * 2);
  _Float16* gg  = (_Float16*)take((size_t)MTOT * FF * 2);
  _Float16* wpq = (_Float16*)take((size_t)DD * DD * 2);
  _Float16* wpk = (_Float16*)take((size_t)DD * DD * 2);
  _Float16* wpv = (_Float16*)take((size_t)DD * DD * 2);
  _Float16* wf1 = (_Float16*)take((size_t)FF * DD * 2);
  _Float16* wf2 = (_Float16*)take((size_t)DD * FF * 2);

  // zero padded head-dim regions read by QK^T (K = 64 includes 16 zero lanes)
  hipMemsetAsync(qpb, 0, (size_t)BB * HH * NN * HDP * 2, stream);
  hipMemsetAsync(kpb, 0, (size_t)BB * HH * KVMIX * HDP * 2, stream);

  // weight conversion f32 -> f16
  {
    int n1 = DD * DD, n2 = FF * DD;
    cvt_f32_f16<<<(n1 + 255) / 256, 256, 0, stream>>>(pq_w, wpq, n1);
    cvt_f32_f16<<<(n1 + 255) / 256, 256, 0, stream>>>(pk_w, wpk, n1);
    cvt_f32_f16<<<(n1 + 255) / 256, 256, 0, stream>>>(pv_w, wpv, n1);
    cvt_f32_f16<<<(n2 + 255) / 256, 256, 0, stream>>>(ff1_w, wf1, n2);
    cvt_f32_f16<<<(n2 + 255) / 256, 256, 0, stream>>>(ff2_w, wf2, n2);
  }

  // LN1
  ln1_kernel<<<MTOT, 128, 0, stream>>>(x, ln1_g, ln1_b, xn);

  // depthwise convs (+BN): q stride1 on both regions, k/v stride2
  {
    int nqs = BB * NS * DD, nqt = BB * NT * DD;
    int nks = BB * 256 * DD, nkt = BB * 64 * DD;
    dwconv_kernel<<<(nqs + 255) / 256, 256, 0, stream>>>(xn, dwq_w, dwq_b, bnq_g, bnq_b, qc, 0, 32, 0, 32, 1, NN, nqs);
    dwconv_kernel<<<(nqt + 255) / 256, 256, 0, stream>>>(xn, dwq_w, dwq_b, bnq_g, bnq_b, qc, NS, 16, NS, 16, 1, NN, nqt);
    dwconv_kernel<<<(nks + 255) / 256, 256, 0, stream>>>(xn, dwk_w, dwk_b, bnk_g, bnk_b, kc, 0, 32, 0, 16, 2, KVMIX, nks);
    dwconv_kernel<<<(nkt + 255) / 256, 256, 0, stream>>>(xn, dwk_w, dwk_b, bnk_g, bnk_b, kc, NS, 16, 256, 8, 2, KVMIX, nkt);
    dwconv_kernel<<<(nks + 255) / 256, 256, 0, stream>>>(xn, dwv_w, dwv_b, bnv_g, bnv_b, vc, 0, 32, 0, 16, 2, KVMIX, nks);
    dwconv_kernel<<<(nkt + 255) / 256, 256, 0, stream>>>(xn, dwv_w, dwv_b, bnv_g, bnv_b, vc, NS, 16, 256, 8, 2, KVMIX, nkt);
  }

  // Q/K/V projections (WMMA). Q pre-scaled by 1/sqrt(HD).
  gemm_proj<<<dim3(MTOT / 128, DD / 64), 256, 0, stream>>>(qc, wpq, pq_b, qpb, NN, 0, SCALE_Q);
  gemm_proj<<<dim3(BB * KVMIX / 128, DD / 64), 256, 0, stream>>>(kc, wpk, pk_b, kpb, KVMIX, 1, 1.0f);
  gemm_proj<<<dim3(BB * KVMIX / 128, DD / 64), 256, 0, stream>>>(vc, wpv, pv_b, vtb, KVMIX, 2, 1.0f);

  // attention: search q (tokens 0..1023) vs mixed kv (0..319); target q vs target kv (256..319)
  attn_kernel<<<dim3(BB * HH, NS / 64), 128, 0, stream>>>(qpb, kpb, vtb, ao, 0, 0, KVMIX);
  attn_kernel<<<dim3(BB * HH, NT / 64), 128, 0, stream>>>(qpb, kpb, vtb, ao, NS, 256, 64);

  // residual + LN2
  ln2_kernel<<<MTOT, 128, 0, stream>>>(x, ao, ln2_g, ln2_b, y, hh);

  // FFN (WMMA): D->FF with GELU, FF->D fused with bias + residual
  gemm_ff1<<<dim3(MTOT / 128, FF / 64), 256, 0, stream>>>(hh, wf1, ff1_b, gg);
  gemm_ff2<<<dim3(MTOT / 128, DD / 64), 256, 0, stream>>>(gg, wf2, ff2_b, y, out);
}
